// MultiHeadAttention_10960756539999
// MI455X (gfx1250) — matrix operand order repaired and checked by reading, not compiled, not run on hardware
//
#include <hip/hip_runtime.h>
#include <hip/hip_bf16.h>

// ---------------- types / helpers ----------------
typedef __attribute__((ext_vector_type(16))) __bf16 v16bf;
typedef __attribute__((ext_vector_type(8)))  float  v8f;

union Frag16 { uint4 u[2]; v16bf v; };

// gfx1250 async global->LDS path (guarded: falls back to sync copy if the
// toolchain doesn't declare the builtin)
#if defined(__has_builtin)
#  if __has_builtin(__builtin_amdgcn_global_load_async_to_lds_b128)
#    define USE_ASYNC_LDS 1
#  endif
#endif

#ifdef USE_ASYNC_LDS
typedef int v4i_t __attribute__((vector_size(16)));
typedef __attribute__((address_space(1))) v4i_t as1_v4i;   // global (__device__)
typedef __attribute__((address_space(3))) v4i_t as3_v4i;   // LDS (__shared__)
#endif

// copy 16 bytes global -> LDS (async when available)
__device__ __forceinline__ void cp16_to_lds(unsigned short* l, const unsigned short* g) {
#ifdef USE_ASYNC_LDS
    __builtin_amdgcn_global_load_async_to_lds_b128(
        (as1_v4i*)g, (as3_v4i*)l, /*offset=*/0, /*cpol=*/0);
#else
    *(uint4*)l = *(const uint4*)g;
#endif
}

__device__ __forceinline__ void wait_async_copies() {
#ifdef USE_ASYNC_LDS
    asm volatile("s_wait_asynccnt 0x0" ::: "memory");
#endif
}

__device__ __forceinline__ unsigned short f32_to_bf16(float f) {
    unsigned int u = __float_as_uint(f);
    u += 0x7FFFu + ((u >> 16) & 1u);   // round-to-nearest-even
    return (unsigned short)(u >> 16);
}

// A-fragment (16x32 bf16, MxK): lane holds row M=lane%16.
// half=lane/16: elems 0..7 = A[M][k0+half*8 .. +7], elems 8..15 = A[M][k0+16+half*8 .. +7]
__device__ __forceinline__ v16bf loadA(const unsigned short* __restrict__ p,
                                       int stride, int k0, int lane) {
    const int row  = lane & 15;
    const int half = lane >> 4;
    const unsigned short* r = p + (size_t)row * stride + k0 + half * 8;
    Frag16 f;
    f.u[0] = *(const uint4*)(r);
    f.u[1] = *(const uint4*)(r + 16);
    return f.v;
}

// B-fragment (32x16 bf16, KxN), source stored [N][K] row-major:
// lane holds column N=lane%16; elems 0..7 = B[half*8 + e][N], elems 8..15 = B[16 + half*8 + (e-8)][N].
__device__ __forceinline__ v16bf loadB(const unsigned short* __restrict__ p,
                                       int stride, int k0, int lane) {
    const int col  = lane & 15;
    const int half = lane >> 4;
    const unsigned short* r = p + (size_t)col * stride + k0 + half * 8;
    Frag16 f;
    f.u[0] = *(const uint4*)(r);
    f.u[1] = *(const uint4*)(r + 16);
    return f.v;
}

__device__ __forceinline__ v8f wmma_bf16(v16bf a, v16bf b, v8f c) {
    return __builtin_amdgcn_wmma_f32_16x16x32_bf16(
        false, a, false, b, (short)0, c, false, false);
}

// ---------------- problem constants ----------------
#define NB   2
#define S    2048
#define NI   768
#define NHD  12
#define DH   64
#define NQKV 2304
#define MTOT (NB * S)          // 4096 rows

// ---------------- kernel 0a: f32 -> bf16 elementwise ----------------
__global__ void k_conv_bf16(const float* __restrict__ x,
                            unsigned short* __restrict__ y, int total) {
    int i = blockIdx.x * blockDim.x + threadIdx.x;
    if (i < total) y[i] = f32_to_bf16(x[i]);
}

// ---------------- kernel 0b: transpose + convert W[K][N] f32 -> Wt[N][K] bf16 ----
__global__ void k_transpose_bf16(const float* __restrict__ W,
                                 unsigned short* __restrict__ Wt,
                                 int Kdim, int Ndim) {
    int i = blockIdx.x * blockDim.x + threadIdx.x;
    int total = Kdim * Ndim;
    if (i >= total) return;
    int nn = i / Kdim;
    int kk = i - nn * Kdim;
    Wt[i] = f32_to_bf16(W[(size_t)kk * Ndim + nn]);
}

// ---------------- kernel 1: QKV GEMM + head-split epilogue -------------------
// C[4096][2304] = A[4096][768] * W[768][2304] + bias ; scatter into Q,K (row-major
// [n,h,s,64]) and Vt (transposed [n,h,64,s]), all bf16.
// Wave tile 32x64 (8 accumulators) with register double-buffered fragments.
__global__ void k_qkv_gemm(const unsigned short* __restrict__ A,   // [4096][768] bf16
                           const unsigned short* __restrict__ Bt,  // [2304][768] bf16
                           const float* __restrict__ bias,         // [2304]
                           unsigned short* __restrict__ Q,
                           unsigned short* __restrict__ K,
                           unsigned short* __restrict__ Vt) {
    const int lane = threadIdx.x & 31;
    const int wave = threadIdx.x >> 5;
    const int m0 = blockIdx.y * 256 + wave * 32;
    const int n0 = blockIdx.x * 64;
    const unsigned short* A0 = A + (size_t)m0 * NI;
    const unsigned short* A1 = A + (size_t)(m0 + 16) * NI;

    v8f acc[2][4] = {};
    v16bf aC[2], bC[4], aN[2], bN[4];
    aC[0] = loadA(A0, NI, 0, lane);
    aC[1] = loadA(A1, NI, 0, lane);
    #pragma unroll
    for (int t = 0; t < 4; ++t)
        bC[t] = loadB(Bt + (size_t)(n0 + t * 16) * NI, NI, 0, lane);

    for (int k = 0; k < NI; k += 32) {
        const int kn = (k + 32 < NI) ? (k + 32) : k;   // dummy reload on last iter
        aN[0] = loadA(A0, NI, kn, lane);
        aN[1] = loadA(A1, NI, kn, lane);
        #pragma unroll
        for (int t = 0; t < 4; ++t)
            bN[t] = loadB(Bt + (size_t)(n0 + t * 16) * NI, NI, kn, lane);
        #pragma unroll
        for (int t = 0; t < 4; ++t) {
            acc[0][t] = wmma_bf16(aC[0], bC[t], acc[0][t]);
            acc[1][t] = wmma_bf16(aC[1], bC[t], acc[1][t]);
        }
        aC[0] = aN[0]; aC[1] = aN[1];
        #pragma unroll
        for (int t = 0; t < 4; ++t) bC[t] = bN[t];
    }

    const int half = lane >> 4, nloc = lane & 15;
    #pragma unroll
    for (int jm = 0; jm < 2; ++jm) {
        #pragma unroll
        for (int t = 0; t < 4; ++t) {
            const int c = n0 + t * 16 + nloc;
            const int h = c / 192;
            const int rem = c - h * 192;
            const float bv = bias[c];
            #pragma unroll
            for (int j = 0; j < 8; ++j) {
                const int r = m0 + jm * 16 + half * 8 + j;
                const int nb = r >> 11, srow = r & (S - 1);
                unsigned short val = f32_to_bf16(acc[jm][t][j] + bv);
                const size_t hoff = (size_t)(nb * NHD + h);
                if (rem < 64)
                    Q[(hoff * S + srow) * DH + rem] = val;
                else if (rem < 128)
                    K[(hoff * S + srow) * DH + (rem - 64)] = val;
                else
                    Vt[(hoff * DH + (rem - 128)) * S + srow] = val;
            }
        }
    }
}

// ---------------- kernel 2: flash attention -------------------
// grid (S/64, NHD, NB), block 128 (4 waves); each wave owns 16 query rows.
// 64-key chunks; K/Vt chunks staged in LDS (shared by all 4 waves), double
// buffered with async global->LDS copies.
__global__ void k_flash_attn(const unsigned short* __restrict__ Q,
                             const unsigned short* __restrict__ Kb,
                             const unsigned short* __restrict__ Vt,
                             unsigned short* __restrict__ ao) {
    __shared__ __align__(16) unsigned short Ktile[2][64][72];  // [buf][key][d]
    __shared__ __align__(16) unsigned short Vtile[2][64][72];  // [buf][d][key]
    __shared__ __align__(16) unsigned short Pbuf[4][16][72];   // per-wave P (16x64)

    const int tid  = threadIdx.x;
    const int lane = tid & 31;
    const int wave = tid >> 5;
    const int h  = blockIdx.y;
    const int nb = blockIdx.z;
    const int q0 = blockIdx.x * 64 + wave * 16;

    const size_t hoff = (size_t)(nb * NHD + h);
    const unsigned short* Qh = Q  + hoff * S * DH;
    const unsigned short* Kh = Kb + hoff * S * DH;
    const unsigned short* Vh = Vt + hoff * DH * S;

    // Q tile A-fragments (d=64 -> two k=32 steps), kept in registers throughout
    const v16bf aq0 = loadA(Qh + (size_t)q0 * DH, DH, 0,  lane);
    const v16bf aq1 = loadA(Qh + (size_t)q0 * DH, DH, 32, lane);

    // stage chunk 0 (512 x 16B units per tile; 4 units per thread per tile)
    #pragma unroll
    for (int c = 0; c < 4; ++c) {
        const int unit = c * 128 + tid;
        const int row = unit >> 3;
        const int u = (unit & 7) * 8;
        cp16_to_lds(&Ktile[0][row][u], Kh + (size_t)row * DH + u);
        cp16_to_lds(&Vtile[0][row][u], Vh + (size_t)row * S + u);
    }
    wait_async_copies();
    __syncthreads();

    v8f o[4] = {};
    float m[8], l[8];
    #pragma unroll
    for (int j = 0; j < 8; ++j) { m[j] = -3.0e38f; l[j] = 0.0f; }
    const int half = lane >> 4, nloc = lane & 15;
    const float inv_scale = 0.125f;  // 1/sqrt(64)

    for (int kc = 0; kc < S; kc += 64) {
        const int cur = (kc >> 6) & 1;
        const int nxt = cur ^ 1;
        if (kc + 64 < S) {  // issue async staging of the next chunk
            const int kc2 = kc + 64;
            #pragma unroll
            for (int c = 0; c < 4; ++c) {
                const int unit = c * 128 + tid;
                const int row = unit >> 3;
                const int u = (unit & 7) * 8;
                cp16_to_lds(&Ktile[nxt][row][u], Kh + (size_t)(kc2 + row) * DH + u);
                cp16_to_lds(&Vtile[nxt][row][u], Vh + (size_t)row * S + kc2 + u);
            }
        }

        // S tile = Q(16x64) * K_chunk^T(64x64) : four 16-col tiles, two k-steps each
        v8f sc[4] = {};
        #pragma unroll
        for (int t = 0; t < 4; ++t) {
            sc[t] = wmma_bf16(aq0, loadB(&Ktile[cur][t * 16][0], 72, 0,  lane), sc[t]);
            sc[t] = wmma_bf16(aq1, loadB(&Ktile[cur][t * 16][0], 72, 32, lane), sc[t]);
        }

        // online softmax (row = half*8 + j per lane; cols across lanes)
        #pragma unroll
        for (int j = 0; j < 8; ++j) {
            float a0 = sc[0][j] * inv_scale;
            float a1 = sc[1][j] * inv_scale;
            float a2 = sc[2][j] * inv_scale;
            float a3 = sc[3][j] * inv_scale;
            float v = fmaxf(fmaxf(a0, a1), fmaxf(a2, a3));
            #pragma unroll
            for (int mk = 1; mk < 16; mk <<= 1)
                v = fmaxf(v, __shfl_xor(v, mk, 16));
            const float mn = fmaxf(m[j], v);
            const float alpha = __expf(m[j] - mn);
            m[j] = mn;
            const float p0 = __expf(a0 - mn);
            const float p1 = __expf(a1 - mn);
            const float p2 = __expf(a2 - mn);
            const float p3 = __expf(a3 - mn);
            float rs = (p0 + p1) + (p2 + p3);
            #pragma unroll
            for (int mk = 1; mk < 16; mk <<= 1)
                rs += __shfl_xor(rs, mk, 16);
            l[j] = l[j] * alpha + rs;
            o[0][j] *= alpha; o[1][j] *= alpha; o[2][j] *= alpha; o[3][j] *= alpha;
            const int row = half * 8 + j;
            Pbuf[wave][row][nloc]      = f32_to_bf16(p0);
            Pbuf[wave][row][16 + nloc] = f32_to_bf16(p1);
            Pbuf[wave][row][32 + nloc] = f32_to_bf16(p2);
            Pbuf[wave][row][48 + nloc] = f32_to_bf16(p3);
        }
        // intra-wave only: LDS ops from one wave execute in order; fence stops
        // compiler reordering of the A-fragment reads above the stores.
        __builtin_amdgcn_fence(__ATOMIC_ACQ_REL, "wavefront");

        // O += P(16x64) * V_chunk(64x64): two k-steps, four 16-col tiles
        const v16bf ap0 = loadA(&Pbuf[wave][0][0], 72, 0,  lane);
        const v16bf ap1 = loadA(&Pbuf[wave][0][0], 72, 32, lane);
        #pragma unroll
        for (int t = 0; t < 4; ++t) {
            o[t] = wmma_bf16(ap0, loadB(&Vtile[cur][t * 16][0], 72, 0,  lane), o[t]);
            o[t] = wmma_bf16(ap1, loadB(&Vtile[cur][t * 16][0], 72, 32, lane), o[t]);
        }

        wait_async_copies();   // our own async stores to buf[nxt] complete
        __syncthreads();       // everyone done reading cur / writing nxt
    }

    // normalize and store merged-head output (bf16) for the proj GEMM
    #pragma unroll
    for (int j = 0; j < 8; ++j) {
        const float inv = 1.0f / l[j];
        const int row = q0 + half * 8 + j;
        #pragma unroll
        for (int t = 0; t < 4; ++t) {
            ao[((size_t)nb * S + row) * NI + h * DH + t * 16 + nloc] =
                f32_to_bf16(o[t][j] * inv);
        }
    }
}

// ---------------- kernel 3: output projection GEMM -------------------
// out[4096][768] = ao[4096][768] * Wproj[768][768] + bproj  (f32 out)
__global__ void k_proj_gemm(const unsigned short* __restrict__ A,   // [4096][768] bf16
                            const unsigned short* __restrict__ Bt,  // [768][768] bf16
                            const float* __restrict__ bias,         // [768]
                            float* __restrict__ out) {
    const int lane = threadIdx.x & 31;
    const int wave = threadIdx.x >> 5;
    const int m0 = blockIdx.y * 256 + wave * 32;
    const int n0 = blockIdx.x * 64;
    const unsigned short* A0 = A + (size_t)m0 * NI;
    const unsigned short* A1 = A + (size_t)(m0 + 16) * NI;

    v8f acc[2][4] = {};
    v16bf aC[2], bC[4], aN[2], bN[4];
    aC[0] = loadA(A0, NI, 0, lane);
    aC[1] = loadA(A1, NI, 0, lane);
    #pragma unroll
    for (int t = 0; t < 4; ++t)
        bC[t] = loadB(Bt + (size_t)(n0 + t * 16) * NI, NI, 0, lane);

    for (int k = 0; k < NI; k += 32) {
        const int kn = (k + 32 < NI) ? (k + 32) : k;
        aN[0] = loadA(A0, NI, kn, lane);
        aN[1] = loadA(A1, NI, kn, lane);
        #pragma unroll
        for (int t = 0; t < 4; ++t)
            bN[t] = loadB(Bt + (size_t)(n0 + t * 16) * NI, NI, kn, lane);
        #pragma unroll
        for (int t = 0; t < 4; ++t) {
            acc[0][t] = wmma_bf16(aC[0], bC[t], acc[0][t]);
            acc[1][t] = wmma_bf16(aC[1], bC[t], acc[1][t]);
        }
        aC[0] = aN[0]; aC[1] = aN[1];
        #pragma unroll
        for (int t = 0; t < 4; ++t) bC[t] = bN[t];
    }

    const int half = lane >> 4, nloc = lane & 15;
    #pragma unroll
    for (int jm = 0; jm < 2; ++jm) {
        #pragma unroll
        for (int t = 0; t < 4; ++t) {
            const int c = n0 + t * 16 + nloc;
            const float bv = bias[c];
            #pragma unroll
            for (int j = 0; j < 8; ++j) {
                const int r = m0 + jm * 16 + half * 8 + j;
                out[(size_t)r * NI + c] = acc[jm][t][j] + bv;
            }
        }
    }
}

// ---------------- launch -------------------
extern "C" void kernel_launch(void* const* d_in, const int* in_sizes, int n_in,
                              void* d_out, int out_size, void* d_ws, size_t ws_size,
                              hipStream_t stream) {
    const float* inp   = (const float*)d_in[0];
    const float* Wqkv  = (const float*)d_in[1];
    const float* bqkv  = (const float*)d_in[2];
    const float* Wproj = (const float*)d_in[3];
    const float* bproj = (const float*)d_in[4];
    float* out = (float*)d_out;

    char* ws = (char*)d_ws;
    size_t off = 0;
    auto alloc = [&](size_t bytes) -> void* {
        void* p = ws + off;
        off = (off + bytes + 255) & ~(size_t)255;
        return p;
    };
    unsigned short* inp_b  = (unsigned short*)alloc((size_t)MTOT * NI * 2);  // reused as ao
    unsigned short* wqkvt  = (unsigned short*)alloc((size_t)NQKV * NI * 2);
    unsigned short* wprojt = (unsigned short*)alloc((size_t)NI * NI * 2);
    unsigned short* Qb     = (unsigned short*)alloc((size_t)NB * NHD * S * DH * 2);
    unsigned short* Kb     = (unsigned short*)alloc((size_t)NB * NHD * S * DH * 2);
    unsigned short* Vtb    = (unsigned short*)alloc((size_t)NB * NHD * DH * S * 2);
    unsigned short* ao     = inp_b;  // safe: inp_b consumed before attention runs

    {
        int tot = MTOT * NI;
        k_conv_bf16<<<(tot + 255) / 256, 256, 0, stream>>>(inp, inp_b, tot);
    }
    {
        int tot = NI * NQKV;
        k_transpose_bf16<<<(tot + 255) / 256, 256, 0, stream>>>(Wqkv, wqkvt, NI, NQKV);
    }
    {
        int tot = NI * NI;
        k_transpose_bf16<<<(tot + 255) / 256, 256, 0, stream>>>(Wproj, wprojt, NI, NI);
    }

    k_qkv_gemm<<<dim3(NQKV / 64, MTOT / 256), 256, 0, stream>>>(
        inp_b, wqkvt, bqkv, Qb, Kb, Vtb);

    k_flash_attn<<<dim3(S / 64, NHD, NB), 128, 0, stream>>>(Qb, Kb, Vtb, ao);

    k_proj_gemm<<<dim3(NI / 64, MTOT / 256), 256, 0, stream>>>(
        ao, wprojt, bproj, out);
}
